// ALSHConvNet_87162066305061
// MI455X (gfx1250) — compile-verified, matrix operand order checked by reading, not run on hardware
//
#include <hip/hip_runtime.h>

typedef __attribute__((ext_vector_type(16))) _Float16 v16h;
typedef __attribute__((ext_vector_type(8)))  float    v8f;

#define ALSH_R 0.2f
#define ALSH_EPS 1e-12f

__device__ __forceinline__ int imin(int a, int b) { return a < b ? a : b; }
__device__ __forceinline__ int imax(int a, int b) { return a > b ? a : b; }

// K-index held by (half h, lane) in the 16-bit A/B fragment layout:
// lanes 0-15: K = {0..7, 16..23}[h]; lanes 16-31: +8
__device__ __forceinline__ int wmma_kmap(int h, int lane) {
    return ((h & 8) ? (h + 8) : h) + ((lane >> 4) << 3);
}

// ---------------------------------------------------------------------------
// ALSH kernel-side hash: kh[co] = floor((P[co].a + c)/R) mod 2
// P = [Kf/(max||Kf||+eps), n^2, n^4, n^8, n^16, n^32]
// ---------------------------------------------------------------------------
__global__ void alsh_kernel_hash(const float* __restrict__ Wt,
                                 const float* __restrict__ a,
                                 const float* __restrict__ c,
                                 int Cout, int K, int* __restrict__ kh) {
    __shared__ float norms[32];
    __shared__ float maxn;
    int t = threadIdx.x;
    if (t < Cout) {
        float s = 0.f;
        for (int k = 0; k < K; ++k) { float w = Wt[t * K + k]; s += w * w; }
        norms[t] = sqrtf(s);
    }
    __syncthreads();
    if (t == 0) {
        float m = 0.f;
        for (int i = 0; i < Cout; ++i) m = fmaxf(m, norms[i]);
        maxn = m + ALSH_EPS;
    }
    __syncthreads();
    if (t < Cout) {
        float inv = 1.0f / maxn;
        float dot = 0.f;
        for (int k = 0; k < K; ++k) dot += Wt[t * K + k] * a[k];
        dot *= inv;
        float nr = norms[t] * inv;
        float p = nr * nr;                 // n^2, n^4, n^8, n^16, n^32
        for (int i = 0; i < 5; ++i) { dot += p * a[K + i]; p = p * p; }
        kh[t] = ((int)floorf((dot + c[0]) / ALSH_R)) & 1;
    }
}

// ---------------------------------------------------------------------------
// ALSH query-side hash per sample. q = repeat(channel_means, 25) so
// Qv.a = (sum_c cm[c]*aSum[c]) / (sqrt(25*sum cm^2)+eps) + 0.5*sum a[K..K+4]
// ---------------------------------------------------------------------------
__global__ void alsh_query_hash(const float* __restrict__ x,
                                const float* __restrict__ a,
                                const float* __restrict__ c,
                                int C, int HW, int* __restrict__ qh) {
    __shared__ float red[128];
    __shared__ float cm[32];
    __shared__ float asum[32];
    int b = blockIdx.x;
    const float* xb = x + (size_t)b * C * HW;
    for (int ch = 0; ch < C; ++ch) {
        float s = 0.f;
        for (int i = threadIdx.x; i < HW; i += blockDim.x) s += xb[ch * HW + i];
        red[threadIdx.x] = s;
        __syncthreads();
        for (int off = 64; off > 0; off >>= 1) {
            if ((int)threadIdx.x < off) red[threadIdx.x] += red[threadIdx.x + off];
            __syncthreads();
        }
        if (threadIdx.x == 0) cm[ch] = red[0] / (float)HW;
        __syncthreads();
    }
    if ((int)threadIdx.x < C) {
        float s = 0.f;
        for (int j = 0; j < 25; ++j) s += a[threadIdx.x * 25 + j];
        asum[threadIdx.x] = s;
    }
    __syncthreads();
    if (threadIdx.x == 0) {
        int K = C * 25;
        float ss = 0.f, dot = 0.f;
        for (int ch = 0; ch < C; ++ch) { ss += cm[ch] * cm[ch]; dot += cm[ch] * asum[ch]; }
        float nq = sqrtf(25.f * ss) + ALSH_EPS;
        float tail = 0.f;
        for (int i = 0; i < 5; ++i) tail += a[K + i];
        float v = dot / nq + 0.5f * tail;
        qh[b] = ((int)floorf((v + c[0]) / ALSH_R)) & 1;
    }
}

// ---------------------------------------------------------------------------
// Pre-pack B (weight) fragments into per-lane WMMA register order, f16,
// zero-padded for kk >= K and n >= Cout.  Layout: bf[nt][ck][lane][16].
// Done once per call; removes all divergence from the conv inner loop.
// ---------------------------------------------------------------------------
__global__ void pack_bfrag(const float* __restrict__ Wt,
                           int Cout, int K, int nChunks, int NT,
                           _Float16* __restrict__ bf) {
    int tid = blockIdx.x * blockDim.x + threadIdx.x;
    int total = NT * nChunks * 32;
    if (tid >= total) return;
    int lane = tid & 31;
    int rest = tid >> 5;
    int ck = rest % nChunks;
    int nt = rest / nChunks;
    int n = (lane & 15) + nt * 16;
#pragma unroll
    for (int h = 0; h < 16; ++h) {
        int kk = ck * 32 + wmma_kmap(h, lane);
        float v = (kk < K && n < Cout) ? Wt[n * K + kk] : 0.f;
        bf[(size_t)tid * 16 + h] = (_Float16)v;
    }
}

// ---------------------------------------------------------------------------
// Fused implicit-GEMM conv(5x5,pad2) + bias + ReLU + ALSH mask + 2x2 maxpool.
// Fully templated on layer geometry so all address math folds to constants.
// One wave = one 4x4 spatial tile (16 rows) x NT*16 couts.
// A gather: clamped addresses + post-load select (no divergent loads);
// B: pre-packed fragments (one 32B vector load per tile per chunk).
// in : [B, Cin, H, W]   out : [B, Cout, H/2, W/2]
// ---------------------------------------------------------------------------
template<int NT, int Cin, int H, int W, int Cout, int K, int NCHUNK>
__global__ void conv_wmma(const float* __restrict__ in,
                          const _Float16* __restrict__ bfrag,
                          const float* __restrict__ bias,
                          const int* __restrict__ kh,
                          const int* __restrict__ qh,
                          float* __restrict__ out,
                          int B) {
    constexpr int tilesX = W >> 2;
    constexpr int tilesPer = tilesX * (H >> 2);
    const int lane = threadIdx.x & 31;
    const int wave = blockIdx.x * (blockDim.x >> 5) + (threadIdx.x >> 5);
    const int b = wave / tilesPer;
    if (b >= B) return;                       // uniform per wave
    const int tile = wave - b * tilesPer;
    const int ty = tile / tilesX, tx = tile - ty * tilesX;

    const int m  = lane & 15;                 // A row: pixel in 4x4 tile
    const int oy = (ty << 2) + (m >> 2);
    const int ox = (tx << 2) + (m & 3);
    const int n0 = lane & 15;                 // B column
    const float* inb = in + (size_t)b * (Cin * H * W);
    const v16h* bfv = (const v16h*)bfrag;     // [nt][ck][lane] -> v16h

    v8f acc0 = {};
    v8f acc1 = {};

    for (int ck = 0; ck < NCHUNK; ++ck) {
        const int kbase = ck * 32;
        // prefetch next chunk's packed weights (global_prefetch_b8)
        if (ck + 1 < NCHUNK)
            __builtin_prefetch((const void*)(bfv + (ck + 1) * 32 + lane), 0, 3);

        __align__(32) _Float16 abuf[16];
#pragma unroll
        for (int h = 0; h < 16; ++h) {
            int kk  = kbase + wmma_kmap(h, lane);
            int kkc = imin(kk, K - 1);
            int cin = kkc / 25;
            int rem = kkc - cin * 25;
            int dy  = rem / 5;
            int dx  = rem - dy * 5;
            int yy  = oy + dy - 2;
            int xx  = ox + dx - 2;
            bool ok = (kk < K) & (yy >= 0) & (yy < H) & (xx >= 0) & (xx < W);
            int yyc = imin(imax(yy, 0), H - 1);
            int xxc = imin(imax(xx, 0), W - 1);
            float raw = inb[(cin * H + yyc) * W + xxc];   // unconditional load
            abuf[h] = (_Float16)(ok ? raw : 0.f);
        }
        v16h af  = *(const v16h*)abuf;
        v16h bf0 = bfv[(size_t)ck * 32 + lane];
        acc0 = __builtin_amdgcn_wmma_f32_16x16x32_f16(false, af, false, bf0,
                                                      (short)0, acc0, false, false);
        if (NT == 2) {
            v16h bf1 = bfv[(size_t)(NCHUNK + ck) * 32 + lane];
            acc1 = __builtin_amdgcn_wmma_f32_16x16x32_f16(false, af, false, bf1,
                                                          (short)0, acc1, false, false);
        }
    }

    // Epilogue: D layout -> lane owns column n0, rows m = 8*(lane>>4)+i.
    // Rows 0-7 = spatial py 0..1 (pool row 0), rows 8-15 = py 2..3 (pool row 1).
    constexpr int Hp = H >> 1, Wp = W >> 1;
    const int qb = qh[b];
    const int pyP = (ty << 1) + (lane >> 4);
    const int pxP = tx << 1;
#pragma unroll
    for (int t = 0; t < NT; ++t) {
        int n = n0 + 16 * t;
        if (n < Cout) {
            union { v8f v; float f[8]; } u;
            u.v = (t == 0) ? acc0 : acc1;
            float bn = bias[n];
            float mv = (kh[n] == qb) ? 1.f : 0.f;
            float v[8];
#pragma unroll
            for (int i = 0; i < 8; ++i) v[i] = fmaxf(u.f[i] + bn, 0.f) * mv;
            float g0 = fmaxf(fmaxf(v[0], v[1]), fmaxf(v[4], v[5])); // px 0-1
            float g1 = fmaxf(fmaxf(v[2], v[3]), fmaxf(v[6], v[7])); // px 2-3
            float* ob = out + ((size_t)(b * Cout + n) * Hp + pyP) * Wp + pxP;
            ob[0] = g0;
            ob[1] = g1;
        }
    }
}

// ---------------------------------------------------------------------------
// FC head: [B,320] @ Wo[10,320]^T + bo.  One wave = 16 samples x 16 cols.
// Wo fragments pre-packed (zero-padded cols 10..15) -> no divergent loads.
// ---------------------------------------------------------------------------
__global__ void fc_wmma(const float* __restrict__ X,
                        const _Float16* __restrict__ bfrag,
                        const float* __restrict__ bo,
                        float* __restrict__ out, int B) {
    const int lane = threadIdx.x & 31;
    const int wave = blockIdx.x * (blockDim.x >> 5) + (threadIdx.x >> 5);
    const int bBase = wave * 16;
    if (bBase >= B) return;
    const int m = lane & 15;
    const int khi = (lane >> 4) << 3;
    const float* xr = X + (size_t)(bBase + m) * 320;
    const v16h* bfv = (const v16h*)bfrag;
    v8f acc = {};
    for (int ck = 0; ck < 10; ++ck) {          // K = 320 = 10 * 32
        __align__(32) _Float16 abuf[16];
#pragma unroll
        for (int h = 0; h < 8; ++h) {          // two contiguous runs of 8
            abuf[h]     = (_Float16)xr[ck * 32 + khi + h];
            abuf[h + 8] = (_Float16)xr[ck * 32 + khi + 16 + h];
        }
        v16h af = *(const v16h*)abuf;
        v16h bf = bfv[(size_t)ck * 32 + lane];
        acc = __builtin_amdgcn_wmma_f32_16x16x32_f16(false, af, false, bf,
                                                     (short)0, acc, false, false);
    }
    union { v8f v; float f[8]; } u;
    u.v = acc;
    int n = lane & 15;
    int mrow = (lane >> 4) * 8;
    if (n < 10) {
#pragma unroll
        for (int i = 0; i < 8; ++i)
            out[(size_t)(bBase + mrow + i) * 10 + n] = u.f[i] + bo[n];
    }
}

// ---------------------------------------------------------------------------
extern "C" void kernel_launch(void* const* d_in, const int* in_sizes, int n_in,
                              void* d_out, int out_size, void* d_ws, size_t ws_size,
                              hipStream_t stream) {
    const float* x  = (const float*)d_in[0];
    const float* W1 = (const float*)d_in[1];
    const float* b1 = (const float*)d_in[2];
    const float* a1 = (const float*)d_in[3];
    const float* c1 = (const float*)d_in[4];
    const float* W2 = (const float*)d_in[5];
    const float* b2 = (const float*)d_in[6];
    const float* a2 = (const float*)d_in[7];
    const float* c2 = (const float*)d_in[8];
    const float* W3 = (const float*)d_in[9];
    const float* b3 = (const float*)d_in[10];
    const float* a3 = (const float*)d_in[11];
    const float* c3 = (const float*)d_in[12];
    const float* Wo = (const float*)d_in[13];
    const float* bo = (const float*)d_in[14];
    float* out = (float*)d_out;

    const int B = in_sizes[0] / (3 * 32 * 32);   // 4096

    // workspace carve-out (everything fits in L2 on MI455X: ~94 MB < 192 MB)
    char* ws = (char*)d_ws;
    size_t off = 0;
    auto carve = [&](size_t bytes) {
        void* p = ws + off;
        off = (off + bytes + 255) & ~((size_t)255);
        return p;
    };
    float* h1  = (float*)carve((size_t)B * 16 * 16 * 16 * sizeof(float));
    float* h2  = (float*)carve((size_t)B * 20 * 8 * 8 * sizeof(float));
    float* h3  = (float*)carve((size_t)B * 320 * sizeof(float));
    int*   kh1 = (int*)carve(32 * sizeof(int));
    int*   kh2 = (int*)carve(32 * sizeof(int));
    int*   kh3 = (int*)carve(32 * sizeof(int));
    int*   qh1 = (int*)carve((size_t)B * sizeof(int));
    int*   qh2 = (int*)carve((size_t)B * sizeof(int));
    int*   qh3 = (int*)carve((size_t)B * sizeof(int));
    _Float16* bf1 = (_Float16*)carve((size_t)1 * 3  * 32 * 16 * sizeof(_Float16));
    _Float16* bf2 = (_Float16*)carve((size_t)2 * 13 * 32 * 16 * sizeof(_Float16));
    _Float16* bf3 = (_Float16*)carve((size_t)2 * 16 * 32 * 16 * sizeof(_Float16));
    _Float16* bfo = (_Float16*)carve((size_t)1 * 10 * 32 * 16 * sizeof(_Float16));

    // ALSH kernel hashes (tiny, exact fp32)
    alsh_kernel_hash<<<1, 32, 0, stream>>>(W1, a1, c1, 16, 75,  kh1);
    alsh_kernel_hash<<<1, 32, 0, stream>>>(W2, a2, c2, 20, 400, kh2);
    alsh_kernel_hash<<<1, 32, 0, stream>>>(W3, a3, c3, 20, 500, kh3);

    // pack weight fragments into WMMA lane order (f16, zero-padded)
    pack_bfrag<<<(1 * 3  * 32 + 255) / 256, 256, 0, stream>>>(W1, 16, 75,  3,  1, bf1);
    pack_bfrag<<<(2 * 13 * 32 + 255) / 256, 256, 0, stream>>>(W2, 20, 400, 13, 2, bf2);
    pack_bfrag<<<(2 * 16 * 32 + 255) / 256, 256, 0, stream>>>(W3, 20, 500, 16, 2, bf3);
    pack_bfrag<<<(1 * 10 * 32 + 255) / 256, 256, 0, stream>>>(Wo, 10, 320, 10, 1, bfo);

    // Layer 1: 32x32 -> pool 16x16, Cout=16 (1 N-tile), K=75 -> 3 chunks
    alsh_query_hash<<<B, 128, 0, stream>>>(x, a1, c1, 3, 32 * 32, qh1);
    conv_wmma<1, 3, 32, 32, 16, 75, 3>
        <<<B * 64 / 4, 128, 0, stream>>>(x, bf1, b1, kh1, qh1, h1, B);

    // Layer 2: 16x16 -> pool 8x8, Cout=20 (2 N-tiles), K=400 -> 13 chunks
    alsh_query_hash<<<B, 128, 0, stream>>>(h1, a2, c2, 16, 16 * 16, qh2);
    conv_wmma<2, 16, 16, 16, 20, 400, 13>
        <<<B * 16 / 4, 128, 0, stream>>>(h1, bf2, b2, kh2, qh2, h2, B);

    // Layer 3: 8x8 -> pool 4x4, Cout=20, K=500 -> 16 chunks
    alsh_query_hash<<<B, 128, 0, stream>>>(h2, a3, c3, 20, 8 * 8, qh3);
    conv_wmma<2, 20, 8, 8, 20, 500, 16>
        <<<B * 4 / 4, 128, 0, stream>>>(h2, bf3, b3, kh3, qh3, h3, B);

    // FC head: 256 waves
    fc_wmma<<<(B / 16 + 3) / 4, 128, 0, stream>>>(h3, bfo, bo, out, B);
}
